// MultiHeadAttention_25099788878431
// MI455X (gfx1250) — matrix operand order repaired and checked by reading, not compiled, not run on hardware
//
#include <hip/hip_runtime.h>

typedef _Float16 v16h __attribute__((ext_vector_type(16)));
typedef _Float16 v8h  __attribute__((ext_vector_type(8)));
typedef _Float16 v4h  __attribute__((ext_vector_type(4)));
typedef float    v8f  __attribute__((ext_vector_type(8)));
typedef int      v4i  __attribute__((ext_vector_type(4)));

constexpr int D_MODEL = 1024;
constexpr int NHEAD   = 16;
constexpr int DK      = 64;
constexpr int BATCH   = 2;
constexpr int SEQ     = 2048;
constexpr int NTOK    = BATCH * SEQ;

#define GLOBAL_AS __attribute__((address_space(1)))
#define LDS_AS    __attribute__((address_space(3)))

#if __has_builtin(__builtin_amdgcn_global_load_async_to_lds_b128)
#define HAVE_ASYNC_LDS 1
#else
#define HAVE_ASYNC_LDS 0
#endif

__device__ __forceinline__ void wait_async_zero() {
#if __has_builtin(__builtin_amdgcn_s_wait_asynccnt)
  __builtin_amdgcn_s_wait_asynccnt(0);
#else
  asm volatile("s_wait_asynccnt 0x0" ::: "memory");
#endif
}

// 16B global -> LDS copy: async (ASYNCcnt-tracked, no VGPR round trip) when
// the gfx1250 builtin is available, else a plain vector copy.
__device__ __forceinline__ void copy_b128_to_lds(const _Float16* g,
                                                 _Float16* l) {
#if HAVE_ASYNC_LDS
  __builtin_amdgcn_global_load_async_to_lds_b128(
      (GLOBAL_AS v4i*)(GLOBAL_AS void*)(_Float16*)g,
      (LDS_AS v4i*)(LDS_AS void*)l, 0, 0);
#else
  *(v8h*)l = *(const v8h*)g;
#endif
}

// ---------------------------------------------------------------------------
// WMMA fragment loaders (CDNA5 v_wmma_f32_16x16x32_f16, wave32)
// A (16x32, f16): lane m = lane&15 holds K {0..7, 16..23} (lanes 0-15)
//                 or K {8..15, 24..31} (lanes 16-31).
// B (32x16, f16): lane n = lane&15 holds K {0..7, 16..23} (lanes 0-15)
//                 or K {8..15, 24..31} (lanes 16-31), matching A.
// C (16x16, f32): VGPR r: M = r (lanes 0-15, N=lane) / M = r+8 (lanes 16-31).
// ---------------------------------------------------------------------------
__device__ __forceinline__ v16h load_a_frag(const _Float16* row_k0, int lane) {
  const _Float16* p = row_k0 + ((lane & 16) ? 8 : 0);
  v8h lo = *(const v8h*)(p);
  v8h hi = *(const v8h*)(p + 16);
  v16h f;
#pragma unroll
  for (int i = 0; i < 8; ++i) { f[i] = lo[i]; f[i + 8] = hi[i]; }
  return f;
}

__device__ __forceinline__ v16h load_b_frag(const _Float16* row_k0, int lane) {
  const _Float16* p = row_k0 + ((lane & 16) ? 8 : 0);
  v8h lo = *(const v8h*)(p);
  v8h hi = *(const v8h*)(p + 16);
  v16h f;
#pragma unroll
  for (int i = 0; i < 8; ++i) { f[i] = lo[i]; f[i + 8] = hi[i]; }
  return f;
}

__device__ __forceinline__ v8f wmma_f16(v16h a, v16h b, v8f c) {
  return __builtin_amdgcn_wmma_f32_16x16x32_f16(false, a, false, b,
                                                (short)0, c, false, false);
}

// ---------------------------------------------------------------------------
// f32 -> f16 conversion (one streaming pass; negligible vs GEMM time)
// ---------------------------------------------------------------------------
__global__ void cvt_f32_f16(const float* __restrict__ src,
                            _Float16* __restrict__ dst, int n) {
  int i = (blockIdx.x * blockDim.x + threadIdx.x) * 4;
  if (i + 3 < n) {
    float4 v = *(const float4*)(src + i);
    v4h o = { (_Float16)v.x, (_Float16)v.y, (_Float16)v.z, (_Float16)v.w };
    *(v4h*)(dst + i) = o;
  }
}

// ---------------------------------------------------------------------------
// C[M,N] = A[M,K] * W[N,K]^T + bias   (row-major; K = N = D_MODEL)
// One wave -> 32x64 output tile: 8 WMMAs per K-step, every B fragment
// reused twice.  8 waves / block share the same 64 W columns.
// MODE 0: f16 row-major out.  MODE 1: f32 row-major out.
// MODE 2: f16 out in V-transposed layout [B, H, DK, SEQ] (for attention).
// ---------------------------------------------------------------------------
template <int MODE>
__global__ __launch_bounds__(256) void gemm_nt(
    const _Float16* __restrict__ A, const _Float16* __restrict__ W,
    const float* __restrict__ bias, _Float16* __restrict__ C16,
    float* __restrict__ C32, int M) {
  constexpr int K = D_MODEL;
  const int lane = threadIdx.x & 31;
  const int wave = threadIdx.x >> 5;
  const int m0 = (blockIdx.x * 8 + wave) * 32;
  const int n0 = blockIdx.y * 64;
  if (m0 >= M) return;

  v8f c[2][4] = {};
  const _Float16* arow0 = A + (size_t)(m0 + (lane & 15)) * K;
  const _Float16* arow1 = arow0 + (size_t)16 * K;
#pragma unroll 2
  for (int k0 = 0; k0 < K; k0 += 32) {
    v16h a0 = load_a_frag(arow0 + k0, lane);
    v16h a1 = load_a_frag(arow1 + k0, lane);
#pragma unroll
    for (int nn = 0; nn < 4; ++nn) {
      const _Float16* wrow = W + (size_t)(n0 + nn * 16 + (lane & 15)) * K + k0;
      v16h b = load_b_frag(wrow, lane);
      c[0][nn] = wmma_f16(a0, b, c[0][nn]);
      c[1][nn] = wmma_f16(a1, b, c[1][nn]);
    }
  }
  const int hl8 = (lane & 16) ? 8 : 0;
#pragma unroll
  for (int mm = 0; mm < 2; ++mm) {
    const int rbase = m0 + mm * 16 + hl8;
#pragma unroll
    for (int nn = 0; nn < 4; ++nn) {
      const int col = n0 + nn * 16 + (lane & 15);
      const float bb = bias[col];
#pragma unroll
      for (int r = 0; r < 8; ++r) {
        const int row = rbase + r;  // token index
        const float v = c[mm][nn][r] + bb;
        if (MODE == 1) {
          C32[(size_t)row * D_MODEL + col] = v;
        } else if (MODE == 0) {
          C16[(size_t)row * D_MODEL + col] = (_Float16)v;
        } else {  // V^T layout: [B, H, DK, SEQ]
          const int bb_ = row >> 11, ss = row & (SEQ - 1);
          const int hh = col >> 6, dd = col & (DK - 1);
          C16[(size_t)((bb_ * NHEAD + hh) * DK + dd) * SEQ + ss] = (_Float16)v;
        }
      }
    }
  }
}

// ---------------------------------------------------------------------------
// Flash attention forward. Block = 256 threads (8 wave32s) handles one
// (batch, head) x 128-query-row tile; each wave owns 16 rows.
// K / V^T tiles are row-contiguous and DOUBLE-BUFFERED in LDS: tile kt+1 is
// filled with GLOBAL_LOAD_ASYNC_TO_LDS_B128 (ASYNCcnt-tracked) while tile kt
// is being consumed by the WMMA pipeline; each wave waits its own ASYNCcnt
// before the barrier, so post-barrier the whole tile is resident.
// ---------------------------------------------------------------------------
__global__ __launch_bounds__(256) void attn_fwd(
    const _Float16* __restrict__ Qh, const _Float16* __restrict__ Kh,
    const _Float16* __restrict__ Vt, const int* __restrict__ q_mask,
    const int* __restrict__ k_mask, const int* __restrict__ causal_p,
    _Float16* __restrict__ O16) {
  __shared__ _Float16 Klds[2][64 * 64];   // K tiles, row-major   (16 KB)
  __shared__ _Float16 VT[2][64 * 64];     // V^T tiles, row-major (16 KB)
  __shared__ _Float16 Plds[8 * 16 * 64];  // per-wave P staging   (16 KB)

  const int tid  = threadIdx.x;
  const int lane = tid & 31;
  const int wave = tid >> 5;
  const int bh = blockIdx.y;
  const int b = bh / NHEAD, h = bh % NHEAD;
  const int hoff = h * DK;
  const int row_base = blockIdx.x * 128;  // query row within sequence
  const int qrow0 = row_base + wave * 16;
  const size_t tok0 = (size_t)b * SEQ;
  const size_t vt0 = (size_t)(b * NHEAD + h) * DK * SEQ;  // V^T row block
  const int causal = causal_p[0];
  const int kt_end = causal ? ((row_base + 127) >> 6) : (SEQ / 64 - 1);
  const int hl8 = (lane & 16) ? 8 : 0;
  const float NEGINF = -__builtin_inff();

  // Q fragments for this wave's 16 rows, d_k split into two K=32 steps
  const _Float16* qrow = Qh + (tok0 + qrow0 + (lane & 15)) * D_MODEL + hoff;
  const v16h qf0 = load_a_frag(qrow, lane);
  const v16h qf1 = load_a_frag(qrow + 32, lane);

  float mrow[8], lrow[8];
#pragma unroll
  for (int r = 0; r < 8; ++r) { mrow[r] = -1e30f; lrow[r] = 0.f; }
  v8f o[4] = {};
  _Float16* pbase = Plds + wave * (16 * 64);

  // Stage one 64-key tile (this thread's 4 16B chunks) into buffer `buf`.
  auto stage_tile = [&](int kt, int buf) {
    const int key0 = kt * 64;
#pragma unroll
    for (int cch = 0; cch < 2; ++cch) {
      const int c = tid + cch * 256;  // 512 16B-chunks per tensor
      const int row = c >> 3, part = (c & 7) * 8;
      copy_b128_to_lds(Kh + (tok0 + key0 + row) * D_MODEL + hoff + part,
                       &Klds[buf][row * 64 + part]);
      copy_b128_to_lds(Vt + vt0 + (size_t)row * SEQ + key0 + part,
                       &VT[buf][row * 64 + part]);
    }
  };

  stage_tile(0, 0);  // prologue fill

  for (int kt = 0; kt <= kt_end; ++kt) {
    const int buf = kt & 1;
    const int key0 = kt * 64;
#if HAVE_ASYNC_LDS
    wait_async_zero();  // this wave's tile-kt chunks have landed
#endif
    __syncthreads();    // whole tile resident; buf^1 readers (kt-1) all done
    if (kt < kt_end) {
      stage_tile(kt + 1, buf ^ 1);  // overlaps with the compute below
      if (kt + 1 < kt_end) {        // pre-warm L2 one stage further ahead
        const int prow = tid >> 2, ppart = (tid & 3) * 16;
        const int key2 = (kt + 2) * 64;
        __builtin_prefetch(Kh + (tok0 + key2 + prow) * D_MODEL + hoff + ppart, 0, 0);
        __builtin_prefetch(Vt + vt0 + (size_t)prow * SEQ + key2 + ppart, 0, 0);
      }
    }

    // Scores: S(16x64) = Q(16x64) * Ktile^T  -> 8 WMMAs
    v8f s[4];
#pragma unroll
    for (int j = 0; j < 4; ++j) {
      const _Float16* krow = &Klds[buf][(j * 16 + (lane & 15)) * 64];
      v16h b0 = load_b_frag(krow, lane);
      v16h b1 = load_b_frag(krow + 32, lane);
      v8f acc = {};
      acc = wmma_f16(qf0, b0, acc);
      acc = wmma_f16(qf1, b1, acc);
      s[j] = acc;
    }
    // Scale (1/sqrt(64)) + causal + key mask
#pragma unroll
    for (int j = 0; j < 4; ++j) {
      const int col = key0 + j * 16 + (lane & 15);
      const int kmv = k_mask[b * SEQ + col];
#pragma unroll
      for (int r = 0; r < 8; ++r) {
        const int myrow = qrow0 + hl8 + r;
        const float v = s[j][r] * 0.125f;
        const bool dead = (kmv == 0) || (causal && (col > myrow));
        s[j][r] = dead ? NEGINF : v;
      }
    }
    // Online softmax (rows live in VGPR r across 16 lanes of each half-wave)
#pragma unroll
    for (int r = 0; r < 8; ++r) {
      float rmax = fmaxf(fmaxf(s[0][r], s[1][r]), fmaxf(s[2][r], s[3][r]));
#pragma unroll
      for (int msk = 1; msk < 16; msk <<= 1)
        rmax = fmaxf(rmax, __shfl_xor(rmax, msk, 32));
      const float mnew = fmaxf(mrow[r], rmax);
      const float alpha = __expf(mrow[r] - mnew);
      mrow[r] = mnew;
      float rsum = 0.f;
#pragma unroll
      for (int j = 0; j < 4; ++j) {
        const float p = __expf(s[j][r] - mnew);
        s[j][r] = p;
        rsum += p;
      }
#pragma unroll
      for (int msk = 1; msk < 16; msk <<= 1) rsum += __shfl_xor(rsum, msk, 32);
      lrow[r] = lrow[r] * alpha + rsum;
#pragma unroll
      for (int nn = 0; nn < 4; ++nn) o[nn][r] *= alpha;
    }
    // P: C-layout -> row-major in LDS -> reload as A fragment (lane transpose)
#pragma unroll
    for (int j = 0; j < 4; ++j) {
#pragma unroll
      for (int r = 0; r < 8; ++r)
        pbase[(hl8 + r) * 64 + j * 16 + (lane & 15)] = (_Float16)s[j][r];
    }
    const _Float16* prow2 = pbase + (lane & 15) * 64;
    v16h pf0 = load_a_frag(prow2, lane);
    v16h pf1 = load_a_frag(prow2 + 32, lane);
    // O += P(16x64) * Vtile(64x64) -> 8 WMMAs (B fragments from V^T rows)
#pragma unroll
    for (int nn = 0; nn < 4; ++nn) {
      const _Float16* vrow = &VT[buf][(nn * 16 + (lane & 15)) * 64];
      v16h vb0 = load_b_frag(vrow, lane);
      v16h vb1 = load_b_frag(vrow + 32, lane);
      o[nn] = wmma_f16(pf0, vb0, o[nn]);
      o[nn] = wmma_f16(pf1, vb1, o[nn]);
    }
  }

  // Epilogue: 1/l normalization, query mask, f16 store for the Wo GEMM.
  float scl[8];
#pragma unroll
  for (int r = 0; r < 8; ++r) {
    const int row = qrow0 + hl8 + r;
    const float inv = (lrow[r] > 0.f) ? (1.f / lrow[r]) : 0.f;
    scl[r] = q_mask[b * SEQ + row] ? inv : 0.f;
  }
#pragma unroll
  for (int nn = 0; nn < 4; ++nn) {
#pragma unroll
    for (int r = 0; r < 8; ++r) {
      const int row = qrow0 + hl8 + r;
      O16[(tok0 + row) * D_MODEL + hoff + nn * 16 + (lane & 15)] =
          (_Float16)(o[nn][r] * scl[r]);
    }
  }
}

// ---------------------------------------------------------------------------
// Host launcher. Workspace (48 MB): q16,k16,Qh,Kh,Vt: 8 MB each;
// Wq16..Wo16: 2 MB each. Attention output reuses q16 (dead after Q proj).
// ---------------------------------------------------------------------------
extern "C" void kernel_launch(void* const* d_in, const int* in_sizes, int n_in,
                              void* d_out, int out_size, void* d_ws,
                              size_t ws_size, hipStream_t stream) {
  const float* q  = (const float*)d_in[0];
  const float* k  = (const float*)d_in[1];
  const int* qm   = (const int*)d_in[2];
  const int* km   = (const int*)d_in[3];
  const float* Wq = (const float*)d_in[4];
  const float* bq = (const float*)d_in[5];
  const float* Wk = (const float*)d_in[6];
  const float* bk = (const float*)d_in[7];
  const float* Wv = (const float*)d_in[8];
  const float* bv = (const float*)d_in[9];
  const float* Wo = (const float*)d_in[10];
  const float* bo = (const float*)d_in[11];
  const int* causal = (const int*)d_in[12];

  char* ws = (char*)d_ws;
  size_t off = 0;
  const size_t tokmat = (size_t)NTOK * D_MODEL * sizeof(_Float16);
  const size_t wmat = (size_t)D_MODEL * D_MODEL * sizeof(_Float16);
  _Float16* q16  = (_Float16*)(ws + off); off += tokmat;
  _Float16* k16  = (_Float16*)(ws + off); off += tokmat;
  _Float16* Qh   = (_Float16*)(ws + off); off += tokmat;
  _Float16* Kh   = (_Float16*)(ws + off); off += tokmat;
  _Float16* Vt   = (_Float16*)(ws + off); off += tokmat;
  _Float16* Wq16 = (_Float16*)(ws + off); off += wmat;
  _Float16* Wk16 = (_Float16*)(ws + off); off += wmat;
  _Float16* Wv16 = (_Float16*)(ws + off); off += wmat;
  _Float16* Wo16 = (_Float16*)(ws + off); off += wmat;
  _Float16* A16  = q16;  // reuse: q16 dead after Q projection

  const int tokN = NTOK * D_MODEL;
  const int wN = D_MODEL * D_MODEL;
  cvt_f32_f16<<<tokN / 4 / 256, 256, 0, stream>>>(q, q16, tokN);
  cvt_f32_f16<<<tokN / 4 / 256, 256, 0, stream>>>(k, k16, tokN);
  cvt_f32_f16<<<wN / 4 / 256, 256, 0, stream>>>(Wq, Wq16, wN);
  cvt_f32_f16<<<wN / 4 / 256, 256, 0, stream>>>(Wk, Wk16, wN);
  cvt_f32_f16<<<wN / 4 / 256, 256, 0, stream>>>(Wv, Wv16, wN);
  cvt_f32_f16<<<wN / 4 / 256, 256, 0, stream>>>(Wo, Wo16, wN);

  dim3 ggrid(NTOK / 256, D_MODEL / 64);
  gemm_nt<0><<<ggrid, 256, 0, stream>>>(q16, Wq16, bq, Qh, nullptr, NTOK);
  gemm_nt<0><<<ggrid, 256, 0, stream>>>(k16, Wk16, bk, Kh, nullptr, NTOK);
  gemm_nt<2><<<ggrid, 256, 0, stream>>>(k16, Wv16, bv, Vt, nullptr, NTOK);

  attn_fwd<<<dim3(SEQ / 128, BATCH * NHEAD), 256, 0, stream>>>(
      Qh, Kh, Vt, qm, km, causal, A16);

  gemm_nt<1><<<ggrid, 256, 0, stream>>>(A16, Wo16, bo, nullptr,
                                        (float*)d_out, NTOK);
}
